// SpectralDistanceAttentionLayer_69535520522572
// MI455X (gfx1250) — matrix operand order repaired and checked by reading, not compiled, not run on hardware
//
#include <hip/hip_runtime.h>

typedef __attribute__((ext_vector_type(16))) _Float16 v16h;
typedef __attribute__((ext_vector_type(8)))  _Float16 v8h;
typedef __attribute__((ext_vector_type(8)))  float    v8f;

#define D_MODEL 256
#define NHEAD   8
#define D_K     32
#define D_FF    1024
#define N_SPEC  16
#define SEQ     2048
#define BATCH   2
#define NEG_SENT (-3.0e38f)

// ---- gfx1250 async load-to-LDS (guarded: falls back to plain copies) ------
#if defined(__AMDGCN__) && __has_builtin(__builtin_amdgcn_global_load_async_to_lds_b128)
#define HAVE_ASYNC_LDS 1
#else
#define HAVE_ASYNC_LDS 0
#endif

#if HAVE_ASYNC_LDS
typedef int v4i __attribute__((ext_vector_type(4)));
typedef __attribute__((address_space(1))) v4i gbl_v4i;   // global b128 operand
typedef __attribute__((address_space(3))) v4i lds_v4i;   // LDS b128 operand

__device__ __forceinline__ gbl_v4i* to_gbl(const void* p) {
    return (gbl_v4i*)(unsigned long long)p;
}
__device__ __forceinline__ lds_v4i* to_lds(const void* p) {
    // generic LDS address: low 32 bits are the LDS byte offset (ISA aperture map)
    return (lds_v4i*)(unsigned)(unsigned long long)p;
}
#endif

__device__ __forceinline__ v8f wmma16(v16h a, v16h b, v8f c) {
    return __builtin_amdgcn_wmma_f32_16x16x32_f16(
        false, a, false, b, (short)0, c, false, false);
}

__device__ __forceinline__ v16h cat8(v8h lo, v8h hi) {
    return __builtin_shufflevector(lo, hi,
        0, 1, 2, 3, 4, 5, 6, 7, 8, 9, 10, 11, 12, 13, 14, 15);
}

// ---------------------------------------------------------------------------
// fp32 -> f16 convert (flat)
// ---------------------------------------------------------------------------
__global__ __launch_bounds__(256) void cvt_f16_kernel(
    const float* __restrict__ in, _Float16* __restrict__ out, int n)
{
    const int i = blockIdx.x * 256 + threadIdx.x;
    if (i < n) out[i] = (_Float16)in[i];
}

// fp32 W[K,N] -> f16 WT[N,K] (transposed for contiguous B fragments)
__global__ __launch_bounds__(256) void cvt_f16_t_kernel(
    const float* __restrict__ W, _Float16* __restrict__ WT, int K, int N)
{
    const int i = blockIdx.x * 256 + threadIdx.x;
    if (i < K * N) {
        const int k = i / N;
        const int n = i - k * N;
        WT[(size_t)n * K + k] = (_Float16)W[i];
    }
}

// ---------------------------------------------------------------------------
// GEMM: C[M,N] = act(A[M,K] @ B[K,N] + bias + res)
// A f16 row-major, BT f16 N-major (transposed). Outputs: fp32 and/or f16
// row-major, and/or V-transposed f16 ([b,h,d,s] layout for attention).
// Block = 256 thr = 8 waves; wave -> 16x16 tile; block -> 16x128.
// ---------------------------------------------------------------------------
__global__ __launch_bounds__(256) void gemm_wmma_f16_kernel(
    const _Float16* __restrict__ A, const _Float16* __restrict__ BT,
    const float* __restrict__ bias, const float* __restrict__ res,
    float* __restrict__ C32, _Float16* __restrict__ C16,
    _Float16* __restrict__ CT16, int M, int N, int K, int relu)
{
    __shared__ alignas(16) _Float16 sA[16][32];   // 1 KB, shared by 8 waves

    const int tid  = threadIdx.x;
    const int lane = tid & 31;
    const int wave = tid >> 5;
    const bool hi  = lane >= 16;
    const int n16  = lane & 15;

    const int nslabs = N >> 7;
    const int mt = blockIdx.x / nslabs;
    const int ns = blockIdx.x % nslabs;
    const int m0 = mt << 4;
    const int n0 = (ns << 7) + (wave << 4);

    v8f acc = {};

    for (int k0 = 0; k0 < K; k0 += 32) {
        __syncthreads();
        {   // cooperative stage of shared A tile (16x32 f16), 2 elems/thread
            const int i0 = tid * 2;
            const int r0 = i0 >> 5, c0 = i0 & 31;
            *(unsigned*)&sA[r0][c0] =
                *(const unsigned*)(A + (size_t)(m0 + r0) * K + k0 + c0);
        }
        __syncthreads();

        // A fragment (contiguous k = 16h + s, matches B): two 16B LDS vector reads
        const v8h alo = *(const v8h*)&sA[n16][hi ? 16 : 0];
        const v8h ahi = *(const v8h*)&sA[n16][hi ? 24 : 8];
        const v16h af = cat8(alo, ahi);

        // B fragment: contiguous 32B from transposed weights
        const v16h bf = *(const v16h*)(
            BT + (size_t)(n0 + n16) * K + k0 + (hi ? 16 : 0));

        acc = wmma16(af, bf, acc);
    }

    // ---- epilogue (hoisted branches) --------------------------------------
    const int n = n0 + n16;
    const int mbase = m0 + (hi ? 8 : 0);
    float vr[8];
    #pragma unroll
    for (int r = 0; r < 8; ++r) vr[r] = acc[r];
    if (bias) {
        const float bz = bias[n];
        #pragma unroll
        for (int r = 0; r < 8; ++r) vr[r] += bz;
    }
    if (res) {
        #pragma unroll
        for (int r = 0; r < 8; ++r) vr[r] += res[(size_t)(mbase + r) * N + n];
    }
    if (relu) {
        #pragma unroll
        for (int r = 0; r < 8; ++r) vr[r] = fmaxf(vr[r], 0.0f);
    }
    if (C32) {
        #pragma unroll
        for (int r = 0; r < 8; ++r) C32[(size_t)(mbase + r) * N + n] = vr[r];
    }
    if (C16) {
        #pragma unroll
        for (int r = 0; r < 8; ++r)
            C16[(size_t)(mbase + r) * N + n] = (_Float16)vr[r];
    }
    if (CT16) {   // V head-transpose: [b, h, d, s]
        const int hd = n >> 5, d = n & 31;
        #pragma unroll
        for (int r = 0; r < 8; ++r) {
            const int mrow = mbase + r;
            const int b = mrow >> 11;            // / SEQ
            const int s = mrow & (SEQ - 1);
            CT16[(((size_t)b * NHEAD + hd) * D_K + d) * SEQ + s] = (_Float16)vr[r];
        }
    }
}

// ---------------------------------------------------------------------------
// Spectral-distance flash attention. Block = 8 waves (one per head) per
// (batch, 16-row q tile); dist/spectral tiles staged ONCE in LDS and shared
// by all heads (spectral_dists is the 512MB roofline limiter).
// ---------------------------------------------------------------------------
__global__ __launch_bounds__(256) void attn_kernel(
    const _Float16* __restrict__ Q, const _Float16* __restrict__ Kp,
    const _Float16* __restrict__ VT, const float* __restrict__ dist,
    const float* __restrict__ spec, const float* __restrict__ Wsp,
    const unsigned char* __restrict__ mask, _Float16* __restrict__ out)
{
    __shared__ alignas(16) float    sDist[16][32];          // 2 KB
    __shared__ alignas(16) float    sSpec[16][32][N_SPEC];  // 32 KB
    __shared__ alignas(16) _Float16 sP[NHEAD][16][32];      // 8 KB

    const int tid  = threadIdx.x;
    const int lane = tid & 31;
    const int h    = tid >> 5;
    const bool hi  = lane >= 16;
    const int n16  = lane & 15;

    const int qt = blockIdx.x % (SEQ / 16);
    const int b  = blockIdx.x / (SEQ / 16);
    const int q0 = qt * 16;
    const size_t rowbase = (size_t)b * SEQ;

    float wspc[N_SPEC];
    #pragma unroll
    for (int c = 0; c < N_SPEC; ++c) wspc[c] = Wsp[c * NHEAD + h];

    // resident Q fragment (contiguous k = 16h + s), two 16B vector loads
    v16h qa;
    {
        const _Float16* qrow = Q + (rowbase + q0 + n16) * D_MODEL + h * D_K;
        qa = cat8(*(const v8h*)(qrow + (hi ? 16 : 0)),
                  *(const v8h*)(qrow + (hi ? 24 : 8)));
    }

    const float scl = 0.17677669529663687f;   // 1/sqrt(32)
    float mrun[8], lrun[8];
    #pragma unroll
    for (int r = 0; r < 8; ++r) { mrun[r] = NEG_SENT; lrun[r] = 0.0f; }
    v8f o0 = {}, o1 = {};

    for (int k0 = 0; k0 < SEQ; k0 += 32) {
        __syncthreads();
        {   // stage dist tile (16x32 fp32)
            const int i0 = tid * 2;
            const int r0 = i0 >> 5, c0 = i0 & 31;
            const float* dp = dist + (rowbase + q0 + r0) * SEQ + k0 + c0;
            sDist[r0][c0]     = dp[0];
            sDist[r0][c0 + 1] = dp[1];
        }
        {   // stage spectral tile (16x32x16 fp32 = 2048 b128), 8 per thread
            #pragma unroll
            for (int j = 0; j < 8; ++j) {
                const int t  = tid * 8 + j;
                const int c4 = t & 3;
                const int kk = (t >> 2) & 31;
                const int qq = t >> 7;
                const float* gp =
                    spec + ((rowbase + q0 + qq) * SEQ + k0 + kk) * N_SPEC + c4 * 4;
#if HAVE_ASYNC_LDS
                __builtin_amdgcn_global_load_async_to_lds_b128(
                    to_gbl(gp), to_lds(&sSpec[qq][kk][c4 * 4]), 0, 0);
#else
                *(float4*)&sSpec[qq][kk][c4 * 4] = *(const float4*)gp;
#endif
                if (k0 + 32 < SEQ)
                    __builtin_prefetch(gp + 32 * N_SPEC, 0, 1);
            }
        }
#if HAVE_ASYNC_LDS
#if __has_builtin(__builtin_amdgcn_s_wait_asynccnt)
        __builtin_amdgcn_s_wait_asynccnt(0);
#else
        asm volatile("s_wait_asynccnt 0" ::: "memory");
#endif
#endif
        __syncthreads();

        // ---- scores: two WMMAs; K fragments are contiguous 32B loads -------
        v8f s0, s1;
        {
            const v8f cz = {};
            const v16h kf0 = *(const v16h*)(
                Kp + (rowbase + k0 + n16) * D_MODEL + h * D_K + (hi ? 16 : 0));
            s0 = wmma16(qa, kf0, cz);
            const v16h kf1 = *(const v16h*)(
                Kp + (rowbase + k0 + 16 + n16) * D_MODEL + h * D_K + (hi ? 16 : 0));
            s1 = wmma16(qa, kf1, cz);
        }

        const unsigned char mk0 = mask[rowbase + k0 + n16];
        const unsigned char mk1 = mask[rowbase + k0 + 16 + n16];

        // ---- scale * dist + spectral bias + mask ---------------------------
        float sc0[8], sc1[8];
        #pragma unroll
        for (int r = 0; r < 8; ++r) {
            const int M = r + (hi ? 8 : 0);
            const float* p0 = &sSpec[M][n16][0];
            const float* p1 = &sSpec[M][16 + n16][0];
            float bia0 = 0.0f, bia1 = 0.0f;
            #pragma unroll
            for (int c = 0; c < N_SPEC; ++c) {
                bia0 += p0[c] * wspc[c];
                bia1 += p1[c] * wspc[c];
            }
            const float v0 = s0[r] * scl * sDist[M][n16]      + bia0;
            const float v1 = s1[r] * scl * sDist[M][16 + n16] + bia1;
            sc0[r] = mk0 ? NEG_SENT : v0;
            sc1[r] = mk1 ? NEG_SENT : v1;
        }

        // ---- online softmax over 16-lane halves of C layout ----------------
        #pragma unroll
        for (int r = 0; r < 8; ++r) {
            const int M = r + (hi ? 8 : 0);
            float mx = fmaxf(sc0[r], sc1[r]);
            #pragma unroll
            for (int off = 1; off < 16; off <<= 1)
                mx = fmaxf(mx, __shfl_xor(mx, off, 32));
            const float mnew = fmaxf(mrun[r], mx);
            const float corr = __expf(mrun[r] - mnew);
            const float p0 = mk0 ? 0.0f : __expf(sc0[r] - mnew);
            const float p1 = mk1 ? 0.0f : __expf(sc1[r] - mnew);
            float rs = p0 + p1;
            #pragma unroll
            for (int off = 1; off < 16; off <<= 1)
                rs += __shfl_xor(rs, off, 32);
            lrun[r] = lrun[r] * corr + rs;
            mrun[r] = mnew;
            o0[r] *= corr;
            o1[r] *= corr;
            sP[h][M][n16]      = (_Float16)p0;   // C->A transpose via LDS
            sP[h][M][16 + n16] = (_Float16)p1;
        }
        asm volatile("s_wait_dscnt 0" ::: "memory");   // wave-local LDS RAW

        const v16h pa = cat8(*(const v8h*)&sP[h][n16][hi ? 16 : 0],
                             *(const v8h*)&sP[h][n16][hi ? 24 : 8]);

        // ---- O += P @ V : V^T rows give contiguous 32B fragments -----------
        {
            const _Float16* vbase =
                VT + (((size_t)b * NHEAD + h) * D_K) * SEQ + k0 + (hi ? 16 : 0);
            const v16h vf0 = *(const v16h*)(vbase + (size_t)n16 * SEQ);
            o0 = wmma16(pa, vf0, o0);
            const v16h vf1 = *(const v16h*)(vbase + (size_t)(16 + n16) * SEQ);
            o1 = wmma16(pa, vf1, o1);
        }
    }

    // ---- normalize + store f16: out[b,q,h*32+d] ---------------------------
    #pragma unroll
    for (int r = 0; r < 8; ++r) {
        const int M = r + (hi ? 8 : 0);
        const float inv = lrun[r] > 0.0f ? 1.0f / lrun[r] : 0.0f;
        const size_t base = (rowbase + q0 + M) * D_MODEL + h * D_K;
        out[base + n16]      = (_Float16)(o0[r] * inv);
        out[base + 16 + n16] = (_Float16)(o1[r] * inv);
    }
}

// ---------------------------------------------------------------------------
// LayerNorm over 256 features; one wave per row; optional f16 copy out.
// ---------------------------------------------------------------------------
__global__ __launch_bounds__(256) void ln_kernel(
    const float* __restrict__ X, const float* __restrict__ g,
    const float* __restrict__ bta, float* __restrict__ Y32,
    _Float16* __restrict__ Y16, int rows)
{
    const int lane = threadIdx.x & 31;
    const int row  = blockIdx.x * 8 + (threadIdx.x >> 5);
    if (row >= rows) return;
    const float* x = X + (size_t)row * D_MODEL;

    float v[8], sum = 0.0f, sq = 0.0f;
    #pragma unroll
    for (int j = 0; j < 8; ++j) {
        v[j] = x[lane + j * 32];
        sum += v[j];
        sq  += v[j] * v[j];
    }
    #pragma unroll
    for (int off = 1; off < 32; off <<= 1) {
        sum += __shfl_xor(sum, off, 32);
        sq  += __shfl_xor(sq,  off, 32);
    }
    const float mean = sum * (1.0f / D_MODEL);
    const float var  = sq * (1.0f / D_MODEL) - mean * mean;
    const float rstd = rsqrtf(var + 1e-5f);
    float o[8];
    #pragma unroll
    for (int j = 0; j < 8; ++j) {
        const int col = lane + j * 32;
        o[j] = (v[j] - mean) * rstd * g[col] + bta[col];
        Y32[(size_t)row * D_MODEL + col] = o[j];
    }
    if (Y16) {
        #pragma unroll
        for (int j = 0; j < 8; ++j)
            Y16[(size_t)row * D_MODEL + lane + j * 32] = (_Float16)o[j];
    }
}

// ---------------------------------------------------------------------------
extern "C" void kernel_launch(void* const* d_in, const int* in_sizes, int n_in,
                              void* d_out, int out_size, void* d_ws, size_t ws_size,
                              hipStream_t stream)
{
    (void)in_sizes; (void)n_in; (void)out_size; (void)ws_size;
    const float* x    = (const float*)d_in[0];
    const float* dist = (const float*)d_in[1];
    const float* spec = (const float*)d_in[2];
    const float* Wq   = (const float*)d_in[3];
    const float* bq   = (const float*)d_in[4];
    const float* Wk   = (const float*)d_in[5];
    const float* bk   = (const float*)d_in[6];
    const float* Wv   = (const float*)d_in[7];
    const float* bv   = (const float*)d_in[8];
    const float* Wo   = (const float*)d_in[9];
    const float* bo   = (const float*)d_in[10];
    const float* Wsp  = (const float*)d_in[11];
    const float* g1   = (const float*)d_in[12];
    const float* b1   = (const float*)d_in[13];
    const float* g2   = (const float*)d_in[14];
    const float* b2   = (const float*)d_in[15];
    const float* W1   = (const float*)d_in[16];
    const float* bf1  = (const float*)d_in[17];
    const float* W2   = (const float*)d_in[18];
    const float* bf2  = (const float*)d_in[19];
    const unsigned char* mask = (const unsigned char*)d_in[20];
    float* out = (float*)d_out;

    const int Mrows = BATCH * SEQ;                     // 4096
    const size_t MB = 1u << 20;
    char* w = (char*)d_ws;
    _Float16* xh  = (_Float16*)(w + 0 * MB);           // 2 MB
    _Float16* Qh  = (_Float16*)(w + 2 * MB);           // 2 MB
    _Float16* Kh  = (_Float16*)(w + 4 * MB);           // 2 MB
    _Float16* VTh = (_Float16*)(w + 6 * MB);           // 2 MB [b,h,d,s]
    _Float16* Ah  = (_Float16*)(w + 8 * MB);           // 2 MB
    _Float16* hh  = (_Float16*)(w + 10 * MB);          // 2 MB
    _Float16* wqh = (_Float16*)(w + 12 * MB);          // 128 KB each:
    _Float16* wkh = wqh + 65536;
    _Float16* wvh = wkh + 65536;
    _Float16* woh = wvh + 65536;
    _Float16* w1h = woh + 65536;                       // 512 KB
    _Float16* w2h = w1h + 262144;                      // 512 KB (ends < 14 MB)
    float*    h32 = (float*)(w + 14 * MB);             // 4 MB
    float*    y32 = (float*)(w + 18 * MB);             // 4 MB
    _Float16* Fh  = (_Float16*)(w + 18 * MB);          // 8 MB (reuses dead y32)
    float*    z32 = (float*)(w + 2 * MB);              // 4 MB (reuses dead Q/K)

    const dim3 blk(256);

    // convert activations + weights to f16 (weights transposed, N-major)
    cvt_f16_kernel<<<4096, blk, 0, stream>>>(x, xh, Mrows * D_MODEL);
    cvt_f16_t_kernel<<<256,  blk, 0, stream>>>(Wq, wqh, D_MODEL, D_MODEL);
    cvt_f16_t_kernel<<<256,  blk, 0, stream>>>(Wk, wkh, D_MODEL, D_MODEL);
    cvt_f16_t_kernel<<<256,  blk, 0, stream>>>(Wv, wvh, D_MODEL, D_MODEL);
    cvt_f16_t_kernel<<<256,  blk, 0, stream>>>(Wo, woh, D_MODEL, D_MODEL);
    cvt_f16_t_kernel<<<1024, blk, 0, stream>>>(W1, w1h, D_MODEL, D_FF);
    cvt_f16_t_kernel<<<1024, blk, 0, stream>>>(W2, w2h, D_FF, D_MODEL);

    const int gD  = (Mrows / 16) * (D_MODEL / 128);    // 512
    const int gFF = (Mrows / 16) * (D_FF   / 128);     // 2048

    gemm_wmma_f16_kernel<<<gD, blk, 0, stream>>>(xh, wqh, bq, nullptr,
        nullptr, Qh, nullptr, Mrows, D_MODEL, D_MODEL, 0);
    gemm_wmma_f16_kernel<<<gD, blk, 0, stream>>>(xh, wkh, bk, nullptr,
        nullptr, Kh, nullptr, Mrows, D_MODEL, D_MODEL, 0);
    gemm_wmma_f16_kernel<<<gD, blk, 0, stream>>>(xh, wvh, bv, nullptr,
        nullptr, nullptr, VTh, Mrows, D_MODEL, D_MODEL, 0);

    attn_kernel<<<BATCH * (SEQ / 16), blk, 0, stream>>>(
        Qh, Kh, VTh, dist, spec, Wsp, mask, Ah);

    gemm_wmma_f16_kernel<<<gD, blk, 0, stream>>>(Ah, woh, bo, x,
        y32, nullptr, nullptr, Mrows, D_MODEL, D_MODEL, 0);
    ln_kernel<<<Mrows / 8, blk, 0, stream>>>(y32, g1, b1, h32, hh, Mrows);

    gemm_wmma_f16_kernel<<<gFF, blk, 0, stream>>>(hh, w1h, bf1, nullptr,
        nullptr, Fh, nullptr, Mrows, D_FF, D_MODEL, 1);
    gemm_wmma_f16_kernel<<<gD, blk, 0, stream>>>(Fh, w2h, bf2, h32,
        z32, nullptr, nullptr, Mrows, D_MODEL, D_FF, 0);
    ln_kernel<<<Mrows / 8, blk, 0, stream>>>(z32, g2, b2, out, nullptr, Mrows);
}
